// CustomGRU_22256520528910
// MI455X (gfx1250) — matrix operand order repaired and checked by reading, not compiled, not run on hardware
//
#include <hip/hip_runtime.h>

// CDNA5 / gfx1250 GRU: bf16 WMMA GEMMs + persistent grid-synced scan with
// LDS-resident recurrent weights.

#define DEVFN __device__ __forceinline__

typedef __bf16 bf16_t;
typedef __attribute__((ext_vector_type(16))) __bf16        v16bf;
typedef __attribute__((ext_vector_type(8)))  float         v8f;
typedef __attribute__((ext_vector_type(4)))  unsigned int  u32x4;
typedef __attribute__((ext_vector_type(4)))  float         f32x4;

union FragU { v16bf v; u32x4 q[2]; };

constexpr int B_ = 64, S_ = 512, I_ = 256, H_ = 1024;
constexpr int M_ = B_ * S_;  // 32768 rows for the batched GEMMs

// ---------------------------------------------------------------------------
// WMMA helpers
// ---------------------------------------------------------------------------
DEVFN v8f wmma_bf16(v16bf a, v16bf b, v8f c) {
  // (neg_a, A, neg_b, B, c_mod, C, reuse_a, reuse_b)
  return __builtin_amdgcn_wmma_f32_16x16x32_bf16(false, a, false, b, (short)0, c,
                                                 false, false);
}

// A-matrix (16x32 bf16) fragment: lane l holds row M = l&15;
// elements 0..7  = K in [koff, koff+8), elements 8..15 = K in [16+koff, +8),
// koff = (l>>4)*8.  Both chunks are contiguous in a row-major source.
DEVFN v16bf load_a_bf16(const bf16_t* rowp, int koff) {
  FragU f;
  f.q[0] = *(const u32x4*)(rowp + koff);
  f.q[1] = *(const u32x4*)(rowp + 16 + koff);
  return f.v;
}

DEVFN v16bf load_a_f32(const float* rowp, int koff) {
  f32x4 a0 = *(const f32x4*)(rowp + koff);
  f32x4 a1 = *(const f32x4*)(rowp + koff + 4);
  f32x4 a2 = *(const f32x4*)(rowp + 16 + koff);
  f32x4 a3 = *(const f32x4*)(rowp + 16 + koff + 4);
  v16bf v;
#pragma unroll
  for (int i = 0; i < 4; ++i) {
    v[i]      = (bf16_t)a0[i];
    v[4 + i]  = (bf16_t)a1[i];
    v[8 + i]  = (bf16_t)a2[i];
    v[12 + i] = (bf16_t)a3[i];
  }
  return v;
}

// B fragments are pre-packed so each lane reads 32 contiguous bytes.
// Works for both global and LDS-resident fragment buffers.
DEVFN v16bf load_b_frag(const bf16_t* fragbase, int lane) {
  FragU f;
  const u32x4* q = (const u32x4*)(fragbase + lane * 16);
  f.q[0] = q[0];
  f.q[1] = q[1];
  return f.v;
}

// ---------------------------------------------------------------------------
// Weight pack: f32 row-major W[K][N] -> bf16 WMMA B fragments.
// Fragment (nt, kb) holds the 32x16 tile B[kb*32 .. +31][nt*16 .. +15].
// Per-lane layout (B-matrix, 16-bit): lane l -> column n = nt*16 + (l&15);
// element j -> K = kb*32 + (l>>4)*8 + j for j<8, kb*32 + 16 + (l>>4)*8 + (j-8) for j>=8.
// ---------------------------------------------------------------------------
__global__ void pack_b_frags(const float* __restrict__ src, bf16_t* __restrict__ dst,
                             int K, int N) {
  int idx = blockIdx.x * blockDim.x + threadIdx.x;
  int total = K * N;
  if (idx >= total) return;
  int j    = idx & 15;
  int lane = (idx >> 4) & 31;
  int f    = idx >> 9;          // 512 elements per fragment
  int kb   = f % (K >> 5);
  int nt   = f / (K >> 5);
  int n = nt * 16 + (lane & 15);
  int k = kb * 32 + ((lane >> 4) << 3) + (j & 7) + ((j >> 3) << 4);
  dst[idx] = (bf16_t)src[(size_t)k * N + n];
}

// ---------------------------------------------------------------------------
// GEMM1: xp_bf16[M, H] = bf16( x_f32[M, I] @ Wi + bi )
// 128 threads = 4 waves; block tile 64x64; wave = one 16-row strip x 4 n-tiles.
// ---------------------------------------------------------------------------
__global__ void __launch_bounds__(128)
gemm_input(const float* __restrict__ x, const bf16_t* __restrict__ WiP,
           const float* __restrict__ bi, bf16_t* __restrict__ xp) {
  const int lane = threadIdx.x & 31;
  const int wave = threadIdx.x >> 5;
  const int l15  = lane & 15;
  const int lh   = lane >> 4;
  const int koff = lh * 8;
  const int n0   = blockIdx.x * 64;
  const int m0   = blockIdx.y * 64 + wave * 16;
  const int row  = m0 + l15;
  const int KB   = I_ >> 5;  // 8

  v8f acc[4] = {};
  for (int kb = 0; kb < KB; ++kb) {
    v16bf a = load_a_f32(x + (size_t)row * I_ + kb * 32, koff);
#pragma unroll
    for (int j = 0; j < 4; ++j) {
      int nt = (n0 >> 4) + j;
      v16bf b = load_b_frag(WiP + ((size_t)nt * KB + kb) * 512, lane);
      acc[j] = wmma_bf16(a, b, acc[j]);
    }
  }
#pragma unroll
  for (int j = 0; j < 4; ++j) {
    int c = n0 + j * 16 + l15;
    float bv = bi[c];
#pragma unroll
    for (int rr = 0; rr < 8; ++rr) {
      int r = m0 + rr + 8 * lh;  // C/D layout: lane half selects +8 rows
      xp[(size_t)r * H_ + c] = (bf16_t)(acc[j][rr] + bv);
    }
  }
}

// ---------------------------------------------------------------------------
// GEMM2: all three gate preactivations in one pass over xp (A reused 3x).
// 12 accumulators (3 gates x 4 n-tiles) per wave.
// ---------------------------------------------------------------------------
__global__ void __launch_bounds__(128)
gemm_gates(const bf16_t* __restrict__ xp,
           const bf16_t* __restrict__ Wz, const bf16_t* __restrict__ Wr,
           const bf16_t* __restrict__ Wn,
           const float* __restrict__ bz, const float* __restrict__ br,
           const float* __restrict__ bn,
           float* __restrict__ oz, float* __restrict__ og, float* __restrict__ on) {
  const bf16_t* W[3] = {Wz, Wr, Wn};
  const float*  Bv[3] = {bz, br, bn};
  float*        O[3] = {oz, og, on};

  const int lane = threadIdx.x & 31;
  const int wave = threadIdx.x >> 5;
  const int l15  = lane & 15;
  const int lh   = lane >> 4;
  const int koff = lh * 8;
  const int n0   = blockIdx.x * 64;
  const int m0   = blockIdx.y * 64 + wave * 16;
  const int row  = m0 + l15;
  const int KB   = H_ >> 5;  // 32

  v8f acc[3][4] = {};
#pragma unroll 2
  for (int kb = 0; kb < KB; ++kb) {
    v16bf a = load_a_bf16(xp + (size_t)row * H_ + kb * 32, koff);
#pragma unroll
    for (int j = 0; j < 4; ++j) {
      size_t fo = ((size_t)((n0 >> 4) + j) * KB + kb) * 512;
#pragma unroll
      for (int g = 0; g < 3; ++g) {
        acc[g][j] = wmma_bf16(a, load_b_frag(W[g] + fo, lane), acc[g][j]);
      }
    }
  }
#pragma unroll
  for (int g = 0; g < 3; ++g) {
#pragma unroll
    for (int j = 0; j < 4; ++j) {
      int c = n0 + j * 16 + l15;
      float bias = Bv[g][c];
#pragma unroll
      for (int rr = 0; rr < 8; ++rr) {
        int r = m0 + rr + 8 * lh;
        O[g][(size_t)r * H_ + c] = acc[g][j][rr] + bias;
      }
    }
  }
}

// ---------------------------------------------------------------------------
// Init: zero h double-buffers (f32 + bf16) and the grid-barrier state.
// Runs every launch so the barrier generation resets across graph replays.
// ---------------------------------------------------------------------------
__global__ void init_state(float* __restrict__ hf, bf16_t* __restrict__ hb,
                           int* __restrict__ bar) {
  int i = blockIdx.x * blockDim.x + threadIdx.x;
  int n = 2 * B_ * H_;
  if (i < n) {
    hf[i] = 0.0f;
    hb[i] = (bf16_t)0.0f;
  }
  if (i < 2) bar[i] = 0;
}

// ---------------------------------------------------------------------------
// Persistent GRU scan. 64 blocks x 128 threads; block owns a 16-column strip,
// wave owns one 16-row batch tile. The block's 96 KB recurrent-weight slice
// (3 gates x 16 cols x K=1024, bf16 fragments) is staged into LDS ONCE and
// reused for all 512 steps -> WMMA B operands come from ds_load_b128.
// One fenced atomic grid barrier per step.
// ---------------------------------------------------------------------------
__global__ void __launch_bounds__(128)
gru_scan(const float* __restrict__ xz, const float* __restrict__ xr,
         const float* __restrict__ xn,
         const bf16_t* __restrict__ Wzh, const bf16_t* __restrict__ Wrh,
         const bf16_t* __restrict__ Wnh,
         float* __restrict__ hf, bf16_t* __restrict__ hb,
         float* __restrict__ out, float* __restrict__ hfinal,
         int* __restrict__ bar_count, int* __restrict__ bar_gen) {
  extern __shared__ char smem_raw[];
  bf16_t* wlds = (bf16_t*)smem_raw;  // [3][KB*512] = 3 x 32 KB

  const int lane = threadIdx.x & 31;
  const int wave = threadIdx.x >> 5;
  const int l15  = lane & 15;
  const int lh   = lane >> 4;
  const int koff = lh * 8;
  const int nt   = blockIdx.x;        // column tile (16 cols)
  const int n0   = nt * 16;
  const int m0   = wave * 16;         // batch tile
  const int brow = m0 + l15;          // batch row this lane streams A for
  const int c    = n0 + l15;          // output column this lane owns
  const int KB   = H_ >> 5;           // 32
  const int HB   = B_ * H_;
  const int GSL  = KB * 512;          // 16384 bf16 per gate slice

  // ---- one-time weight stage: global -> LDS (3 x 32 KB contiguous) ----
  {
    const bf16_t* gsrc[3] = {Wzh + (size_t)nt * GSL, Wrh + (size_t)nt * GSL,
                             Wnh + (size_t)nt * GSL};
#pragma unroll
    for (int g = 0; g < 3; ++g) {
      const u32x4* s = (const u32x4*)gsrc[g];
      u32x4* d = (u32x4*)(wlds + g * GSL);
      for (int i = threadIdx.x; i < GSL / 8; i += 128) d[i] = s[i];
    }
  }
  __syncthreads();

  const bf16_t* wz_l = wlds;
  const bf16_t* wr_l = wlds + GSL;
  const bf16_t* wn_l = wlds + 2 * GSL;

#pragma unroll 1
  for (int t = 0; t < S_; ++t) {
    const int p = t & 1;
    const bf16_t* hcur = hb + (size_t)p * HB;

    // Prefetch combine operands early so global latency hides under WMMAs.
    float xzv[8], xrv[8], xnv[8];
#pragma unroll
    for (int rr = 0; rr < 8; ++rr) {
      int b = m0 + rr + 8 * lh;
      size_t mrow = (size_t)b * S_ + t;
      xzv[rr] = xz[mrow * H_ + c];
      xrv[rr] = xr[mrow * H_ + c];
      xnv[rr] = xn[mrow * H_ + c];
    }

    v8f az = {}, ar = {}, an = {};
#pragma unroll 4
    for (int kb = 0; kb < KB; ++kb) {
      v16bf a = load_a_bf16(hcur + (size_t)brow * H_ + kb * 32, koff);
      az = wmma_bf16(a, load_b_frag(wz_l + kb * 512, lane), az);
      ar = wmma_bf16(a, load_b_frag(wr_l + kb * 512, lane), ar);
      an = wmma_bf16(a, load_b_frag(wn_l + kb * 512, lane), an);
    }

    const float* hfc = hf + (size_t)p * HB;
    float*  hfn = hf + (size_t)(1 - p) * HB;
    bf16_t* hbn = hb + (size_t)(1 - p) * HB;
#pragma unroll
    for (int rr = 0; rr < 8; ++rr) {
      int b = m0 + rr + 8 * lh;
      size_t mrow = (size_t)b * S_ + t;
      float z  = 1.0f / (1.0f + __expf(-(xzv[rr] + az[rr])));
      float r  = 1.0f / (1.0f + __expf(-(xrv[rr] + ar[rr])));
      float nn = tanhf(xnv[rr] + r * an[rr]);
      float hv = hfc[(size_t)b * H_ + c];
      float hnew = (1.0f - z) * nn + z * hv;
      hfn[(size_t)b * H_ + c] = hnew;
      hbn[(size_t)b * H_ + c] = (bf16_t)hnew;
      out[mrow * H_ + c] = hnew;
      if (t == S_ - 1) hfinal[(size_t)b * H_ + c] = hnew;
    }

    // ---- device-wide barrier (one per step) ----
    __syncthreads();
    if (threadIdx.x == 0) {
      __threadfence();  // release all h writes
      if (atomicAdd(bar_count, 1) == (int)gridDim.x - 1) {
        atomicExch(bar_count, 0);
        __threadfence();
        atomicAdd(bar_gen, 1);
      } else {
        while (atomicAdd(bar_gen, 0) <= t) __builtin_amdgcn_s_sleep(1);
      }
      __threadfence();  // acquire
    }
    __syncthreads();
  }
}

// ---------------------------------------------------------------------------
// Host launch
// ---------------------------------------------------------------------------
extern "C" void kernel_launch(void* const* d_in, const int* in_sizes, int n_in,
                              void* d_out, int out_size, void* d_ws, size_t ws_size,
                              hipStream_t stream) {
  (void)in_sizes; (void)n_in; (void)out_size; (void)ws_size;

  const float* x  = (const float*)d_in[0];
  const float* Wi = (const float*)d_in[1];
  const float* bi = (const float*)d_in[2];
  const float* Wz = (const float*)d_in[3];
  const float* bz = (const float*)d_in[4];
  const float* Wr = (const float*)d_in[5];
  const float* br = (const float*)d_in[6];
  const float* Wn = (const float*)d_in[7];
  const float* bn = (const float*)d_in[8];

  char* ws = (char*)d_ws;
  size_t off = 0;
  auto carve = [&](size_t bytes) -> char* {
    char* p = ws + off;
    off += (bytes + 255) & ~(size_t)255;
    return p;
  };

  bf16_t* WiP  = (bf16_t*)carve((size_t)I_ * H_ * 2);
  bf16_t* WzxP = (bf16_t*)carve((size_t)H_ * H_ * 2);
  bf16_t* WrxP = (bf16_t*)carve((size_t)H_ * H_ * 2);
  bf16_t* WnxP = (bf16_t*)carve((size_t)H_ * H_ * 2);
  bf16_t* WzhP = (bf16_t*)carve((size_t)H_ * H_ * 2);
  bf16_t* WrhP = (bf16_t*)carve((size_t)H_ * H_ * 2);
  bf16_t* WnhP = (bf16_t*)carve((size_t)H_ * H_ * 2);
  bf16_t* xp   = (bf16_t*)carve((size_t)M_ * H_ * 2);
  float*  xzb  = (float*)carve((size_t)M_ * H_ * 4);
  float*  xrb  = (float*)carve((size_t)M_ * H_ * 4);
  float*  xnb  = (float*)carve((size_t)M_ * H_ * 4);
  float*  hf   = (float*)carve((size_t)2 * B_ * H_ * 4);
  bf16_t* hb   = (bf16_t*)carve((size_t)2 * B_ * H_ * 2);
  int*    bar  = (int*)carve(256);

  // Pack all weights into bf16 WMMA B-fragment layout.
  {
    const int thr = 256;
    int nIH = I_ * H_;
    pack_b_frags<<<(nIH + thr - 1) / thr, thr, 0, stream>>>(Wi, WiP, I_, H_);
    int nHH = H_ * H_;
    int blk = (nHH + thr - 1) / thr;
    pack_b_frags<<<blk, thr, 0, stream>>>(Wz,                    WzxP, H_, H_);
    pack_b_frags<<<blk, thr, 0, stream>>>(Wz + (size_t)H_ * H_,  WzhP, H_, H_);
    pack_b_frags<<<blk, thr, 0, stream>>>(Wr,                    WrxP, H_, H_);
    pack_b_frags<<<blk, thr, 0, stream>>>(Wr + (size_t)H_ * H_,  WrhP, H_, H_);
    pack_b_frags<<<blk, thr, 0, stream>>>(Wn,                    WnxP, H_, H_);
    pack_b_frags<<<blk, thr, 0, stream>>>(Wn + (size_t)H_ * H_,  WnhP, H_, H_);
  }

  init_state<<<(2 * B_ * H_ + 255) / 256, 256, 0, stream>>>(hf, hb, bar);

  gemm_input<<<dim3(H_ / 64, M_ / 64), 128, 0, stream>>>(x, WiP, bi, xp);

  gemm_gates<<<dim3(H_ / 64, M_ / 64), 128, 0, stream>>>(
      xp, WzxP, WrxP, WnxP, bz, br, bn, xzb, xrb, xnb);

  float* out    = (float*)d_out;
  float* hfinal = out + (size_t)M_ * H_;
  // 96 KB dynamic LDS: 3 gate slices x 32 KB (CDNA5 WGP has 320 KB LDS).
  gru_scan<<<dim3(H_ / 16), 128, 3 * 32 * 1024, stream>>>(
      xzb, xrb, xnb, WzhP, WrhP, WnhP, hf, hb, out, hfinal, bar, bar + 1);
}
